// AttentionHead_70635032150361
// MI455X (gfx1250) — matrix operand order repaired and checked by reading, not compiled, not run on hardware
//
#include <hip/hip_runtime.h>

typedef __attribute__((ext_vector_type(16))) _Float16 v16h;
typedef __attribute__((ext_vector_type(8)))  _Float16 v8h;
typedef __attribute__((ext_vector_type(8)))  float    v8f;
typedef __attribute__((ext_vector_type(4)))  float    v4f;

union frag16 { v16h v; v8h h[2]; };

#define BB 4
#define TT 4096
#define CC 1024
#define HH 64
#define NFRAG 12   // 3 weights x 4 n-tiles

// Wpack fragment layout: frag f = ((w*4 + nt)*32 + kc), 32 lanes x 16 halves.
// Lane's halves: b[i] = W[nt*16 + lane%16][kc*32 + (lane>>4)*16 + i]

// ---------------------------------------------------------------------------
// Kernel 0: pack Wq/Wk/Wv (fp32 [64,1024]) into f16 B-fragment layout (384KB).
// ---------------------------------------------------------------------------
__global__ __launch_bounds__(256)
void pack_w_kernel(const float* __restrict__ Wq,
                   const float* __restrict__ Wk,
                   const float* __restrict__ Wv,
                   _Float16* __restrict__ Wpack)
{
    const int tid = blockIdx.x * 256 + threadIdx.x;     // 0..12287
    if (tid >= 3 * 4 * 32 * 32) return;
    const int lane = tid & 31;
    const int f    = tid >> 5;          // ((w*4+nt)*32 + kc)
    const int kc   = f & 31;
    const int wnt  = f >> 5;
    const int nt   = wnt & 3;
    const int w    = wnt >> 2;
    const float* W = (w == 0) ? Wq : (w == 1) ? Wk : Wv;
    const float* src = W + (size_t)(nt * 16 + (lane & 15)) * CC
                         + kc * 32 + (lane >> 4) * 16;
    v4f s0 = *(const v4f*)(src);
    v4f s1 = *(const v4f*)(src + 4);
    v4f s2 = *(const v4f*)(src + 8);
    v4f s3 = *(const v4f*)(src + 12);
    v8h d0, d1;
#pragma unroll
    for (int i = 0; i < 4; ++i) {
        d0[i]     = (_Float16)s0[i];
        d0[4 + i] = (_Float16)s1[i];
        d1[i]     = (_Float16)s2[i];
        d1[4 + i] = (_Float16)s3[i];
    }
    _Float16* dst = Wpack + (size_t)tid * 16;
    *(v8h*)(dst)     = d0;
    *(v8h*)(dst + 8) = d1;
}

// ---------------------------------------------------------------------------
// Kernel 1: Q/K/V projection with async-to-LDS double-buffered W staging.
// One wave computes a 16-row M tile x (3 weights x 4 N-tiles).  The block
// cooperatively DMAs the 12KB Wpack slice for k-chunk kc+1 into LDS while
// computing WMMAs on slice kc.  The LDS buffer is written only by the async
// DMA (invisible stores), so its address is escaped into an asm once: after
// that, the "memory"-clobbering async/wait asms are potential writers and
// the (non-volatile) ds_load_b128 fragment reads stay alive and ordered.
// ---------------------------------------------------------------------------
__global__ __launch_bounds__(256)
void qkv_proj_kernel(const float* __restrict__ X,
                     const _Float16* __restrict__ Wpack,
                     _Float16* __restrict__ Qf,
                     _Float16* __restrict__ Kf,
                     _Float16* __restrict__ Vt)
{
    // 2 buffers x 12 frags x 32 lanes x 16 halves = 24 KB
    __shared__ __align__(16) _Float16 wbuf[2][NFRAG * 32 * 16];

    // Escape the LDS buffer so the async-DMA asm counts as a writer to it.
    {
        _Float16* esc = &wbuf[0][0];
        asm volatile("" : "+v"(esc) :: "memory");
    }

    const int lane  = threadIdx.x & 31;
    const int wave  = threadIdx.x >> 5;
    const int col16 = lane & 15;
    const int hi    = lane >> 4;
    const int m0    = (blockIdx.x * 8 + wave) * 16;   // row tile in flat B*T

    // Stage one 12KB slice (kc) into wbuf[bufi]: 768 x 16B units, 3 per thread.
    auto stage = [&](int kc, int bufi) {
#pragma unroll
        for (int j = 0; j < 3; ++j) {
            const int u     = threadIdx.x + j * 256;   // 0..767
            const int wnt   = u >> 6;                  // fragment 0..11
            const int rem   = u & 63;
            const int lane2 = rem >> 1;
            const int hsel  = rem & 1;
            const _Float16* g =
                Wpack + ((size_t)(wnt * 32 + kc) * 32 + lane2) * 16 + hsel * 8;
            const unsigned lds_off =
                (unsigned)((const char*)&wbuf[bufi][u * 8] - (const char*)&wbuf[0][0]);
            asm volatile("global_load_async_to_lds_b128 %0, %1, off"
                         :: "v"(lds_off), "v"(g) : "memory");
        }
    };

    v8f zero = {};
    v8f acc[NFRAG];
#pragma unroll
    for (int f = 0; f < NFRAG; ++f) acc[f] = zero;

    const float* xrow = X + (size_t)(m0 + col16) * CC;   // A row = lane%16

    stage(0, 0);
    asm volatile("s_wait_asynccnt 0" ::: "memory");
    __syncthreads();

#pragma unroll 1
    for (int kc = 0; kc < 32; ++kc) {
        const int cur = kc & 1;
        if (kc + 1 < 32) stage(kc + 1, cur ^ 1);   // overlap DMA with compute

        // ---- A fragment (fp32 -> f16): low lanes K=k0+{0..15}, hi +16
        v16h a;
        {
            const int o1 = kc * 32 + (hi ? 16 : 0);
            v4f x0 = *(const v4f*)(xrow + o1);
            v4f x1 = *(const v4f*)(xrow + o1 + 4);
            v4f x2 = *(const v4f*)(xrow + o1 + 8);
            v4f x3 = *(const v4f*)(xrow + o1 + 12);
#pragma unroll
            for (int i = 0; i < 4; ++i) {
                a[i]      = (_Float16)x0[i];
                a[4 + i]  = (_Float16)x1[i];
                a[8 + i]  = (_Float16)x2[i];
                a[12 + i] = (_Float16)x3[i];
            }
        }

        // ---- 12 B fragments from LDS (ds_load_b128 pairs) + WMMA ----
#pragma unroll
        for (int f = 0; f < NFRAG; ++f) {
            const _Float16* wp = &wbuf[cur][(f * 32 + lane) * 16];
            frag16 bf;
            bf.h[0] = *(const v8h*)(wp);
            bf.h[1] = *(const v8h*)(wp + 8);
            acc[f] = __builtin_amdgcn_wmma_f32_16x16x32_f16(
                false, a, false, bf.v, (short)0, acc[f], false, false);
        }

        asm volatile("s_wait_asynccnt 0" ::: "memory");
        __syncthreads();
    }

    // ---- store: C layout row = r + 8*hi, col = nt*16 + lane%16 ----
#pragma unroll
    for (int nt = 0; nt < 4; ++nt) {
#pragma unroll
        for (int r = 0; r < 8; ++r) {
            const int row = m0 + r + hi * 8;
            const int col = nt * 16 + col16;
            Qf[(size_t)row * HH + col] = (_Float16)acc[0 * 4 + nt][r];
            Kf[(size_t)row * HH + col] = (_Float16)acc[1 * 4 + nt][r];
            const int bb = row >> 12;
            const int t  = row & (TT - 1);
            Vt[((size_t)bb * HH + col) * TT + t] = (_Float16)acc[2 * 4 + nt][r];
        }
    }
}

// ---------------------------------------------------------------------------
// Kernel 2: flash attention. One wave per 16-query tile; streams 32-key
// strips with online softmax.  Complement pairing balances causal work:
// block b owns q-tiles {b*4+w} and {1023-(b*4+w)} -> ~constant strips/block.
// ---------------------------------------------------------------------------
__global__ __launch_bounds__(256)
void flash_attn_kernel(const _Float16* __restrict__ Qf,
                       const _Float16* __restrict__ Kf,
                       const _Float16* __restrict__ Vt,
                       const int* __restrict__ kmask,
                       float* __restrict__ Out)
{
    __shared__ __align__(16) _Float16 plds[8][16 * 32];  // per-wave P tile

    const int lane  = threadIdx.x & 31;
    const int wave  = threadIdx.x >> 5;
    const int col16 = lane & 15;
    const int hi    = lane >> 4;

    // balanced causal mapping
    const int base = blockIdx.x * 4 + (wave & 3);
    const int qt   = (wave < 4) ? base : (1023 - base);
    const int b  = qt >> 8;                 // 256 query tiles per batch
    const int q0 = (qt & 255) * 16;

    // Q A-fragments: two 16x32 chunks over head dim
    frag16 qa[2];
    {
        const _Float16* qrow = Qf + (size_t)(b * TT + q0 + col16) * HH;
#pragma unroll
        for (int c = 0; c < 2; ++c) {
            const int o1 = c * 32 + (hi ? 16 : 0);
            qa[c].h[0] = *(const v8h*)(qrow + o1);
            qa[c].h[1] = *(const v8h*)(qrow + o1 + 8);
        }
    }

    v8f zero = {};
    v8f o[4] = {zero, zero, zero, zero};
    float m[8], l[8];
#pragma unroll
    for (int r = 0; r < 8; ++r) { m[r] = -1.0e30f; l[r] = 0.0f; }

    const int sEnd = (q0 + 15) >> 5;        // inclusive last 32-key strip

#pragma unroll 1
    for (int s = 0; s <= sEnd; ++s) {
        const int kb = s * 32;

        if (s < sEnd) {  // prefetch next strip's K rows / V columns
            __builtin_prefetch(Kf + (size_t)(b * TT + kb + 32 + lane) * HH, 0, 1);
            __builtin_prefetch(Vt + ((size_t)b * HH + lane * 2) * TT + kb + 32, 0, 1);
        }

        // ---- S = Q * K^T for two 16-key tiles ----
        float sv[2][8];
#pragma unroll
        for (int t = 0; t < 2; ++t) {
            const _Float16* krow =
                Kf + (size_t)(b * TT + kb + t * 16 + col16) * HH + hi * 16;
            v8f sacc = zero;
#pragma unroll
            for (int c = 0; c < 2; ++c) {
                frag16 kf;
                kf.h[0] = *(const v8h*)(krow + c * 32);
                kf.h[1] = *(const v8h*)(krow + c * 32 + 8);
                sacc = __builtin_amdgcn_wmma_f32_16x16x32_f16(
                    false, qa[c].v, false, kf.v, (short)0, sacc, false, false);
            }
            const int colg = kb + t * 16 + col16;
            const int km   = kmask[b * TT + colg];
#pragma unroll
            for (int r = 0; r < 8; ++r) {
                const int rowg = q0 + r + hi * 8;
                const float v  = sacc[r] * 0.125f;   // 1/sqrt(64)
                sv[t][r] = (colg <= rowg && km != 0) ? v : -1.0e30f;
            }
        }

        // ---- online softmax (row reductions within 16-lane groups) ----
        float scl[8];
#pragma unroll
        for (int r = 0; r < 8; ++r) {
            float mx = fmaxf(sv[0][r], sv[1][r]);
#pragma unroll
            for (int off = 8; off >= 1; off >>= 1)
                mx = fmaxf(mx, __shfl_xor(mx, off, 32));
            const float mnew = fmaxf(m[r], mx);
            scl[r] = __expf(m[r] - mnew);
            const float p0 = __expf(sv[0][r] - mnew);
            const float p1 = __expf(sv[1][r] - mnew);
            float rs = p0 + p1;
#pragma unroll
            for (int off = 8; off >= 1; off >>= 1)
                rs += __shfl_xor(rs, off, 32);
            l[r] = l[r] * scl[r] + rs;
            m[r] = mnew;
            plds[wave][(r + hi * 8) * 32 + col16]      = (_Float16)p0;
            plds[wave][(r + hi * 8) * 32 + 16 + col16] = (_Float16)p1;
        }
#pragma unroll
        for (int nt = 0; nt < 4; ++nt)
#pragma unroll
            for (int r = 0; r < 8; ++r)
                o[nt][r] *= scl[r];

        // ---- read P back as 16x32 A-fragment ----
        frag16 pa;
        {
            const _Float16* pr = &plds[wave][col16 * 32 + (hi ? 16 : 0)];
            pa.h[0] = *(const v8h*)(pr);
            pa.h[1] = *(const v8h*)(pr + 8);
        }

        // ---- O += P * V (B[k][n] = V[kb+k][n] = Vt[b][n][kb+k]) ----
#pragma unroll
        for (int nt = 0; nt < 4; ++nt) {
            const _Float16* vrow =
                Vt + ((size_t)b * HH + nt * 16 + col16) * TT + kb + hi * 16;
            frag16 vf;
            vf.h[0] = *(const v8h*)(vrow);
            vf.h[1] = *(const v8h*)(vrow + 8);
            o[nt] = __builtin_amdgcn_wmma_f32_16x16x32_f16(
                false, pa.v, false, vf.v, (short)0, o[nt], false, false);
        }
    }

    // ---- epilogue: divide by row sums, store fp32 ----
#pragma unroll
    for (int nt = 0; nt < 4; ++nt)
#pragma unroll
        for (int r = 0; r < 8; ++r) {
            const int rowg = q0 + r + hi * 8;
            Out[(size_t)(b * TT + rowg) * HH + nt * 16 + col16] = o[nt][r] / l[r];
        }
}

// ---------------------------------------------------------------------------
extern "C" void kernel_launch(void* const* d_in, const int* in_sizes, int n_in,
                              void* d_out, int out_size, void* d_ws, size_t ws_size,
                              hipStream_t stream) {
    const float* X  = (const float*)d_in[0];
    const float* Wq = (const float*)d_in[1];
    const float* Wk = (const float*)d_in[2];
    const float* Wv = (const float*)d_in[3];
    const int* km   = (const int*)d_in[4];
    float* out      = (float*)d_out;

    _Float16* Qf    = (_Float16*)d_ws;
    _Float16* Kf    = Qf + (size_t)BB * TT * HH;
    _Float16* Vt    = Kf + (size_t)BB * TT * HH;
    _Float16* Wpack = Vt + (size_t)BB * TT * HH;   // 384 KB

    pack_w_kernel<<<48, 256, 0, stream>>>(Wq, Wk, Wv, Wpack);
    qkv_proj_kernel<<<128, 256, 0, stream>>>(X, Wpack, Qf, Kf, Vt);
    flash_attn_kernel<<<128, 256, 0, stream>>>(Qf, Kf, Vt, km, out);
}
